// ACORPolicy_6897717478056
// MI455X (gfx1250) — compile-verified, matrix operand order checked by reading, not run on hardware
//
#include <hip/hip_runtime.h>
#include <hip/hip_bf16.h>
#include <math.h>

// ---------------------------------------------------------------------------
// ACOR policy forward for MI455X (gfx1250, wave32, WMMA).
// Dense layers: v_wmma_f32_16x16x32_f16 with operands PRE-PACKED into the
// ISA fragment layout (one contiguous 32B load per lane per fragment).
// Each GEMM wave computes a 16x64 tile: 1 A-frag load feeds 4 WMMAs.
// ---------------------------------------------------------------------------

#define BB  32
#define AA  256
#define M8  (BB * AA)      // 8192 tokens
#define KN  16             // K_NEIGH
#define LK  8              // LEADER_K
#define HW  8              // HIST_W
#define EPSW 1e-8f

typedef __attribute__((ext_vector_type(16))) _Float16 v16h;
typedef __attribute__((ext_vector_type(8)))  float    v8f;

__device__ __forceinline__ float gelu_f(float x) {
  return 0.5f * x * (1.0f + erff(x * 0.70710678118654752f));
}
__device__ __forceinline__ float sigmoid_f(float x) {
  return 1.0f / (1.0f + expf(-x));
}
__device__ __forceinline__ float wave_sum(float v) {
#pragma unroll
  for (int m = 16; m >= 1; m >>= 1) v += __shfl_xor(v, m, 32);
  return v;
}

// ---------------------------------------------------------------------------
// Fragment packing. WMMA 16-bit 16x32 layout per 16x16(x32) tile:
//   lane l<16 : row/col = l,   K offsets {0..7, 16..23}
//   lane l>=16: row/col = l-16,K offsets {8..15,24..31}
// Packed element address: ((tile*(K/32) + kk)*32 + lane)*16 + i
// so each lane's 16-half fragment is one contiguous 32-byte chunk.
// ---------------------------------------------------------------------------
__global__ void pack_a_kernel(const float* __restrict__ A,
                              _Float16* __restrict__ Ap,
                              int K, long total) {
  long t = (long)blockIdx.x * blockDim.x + threadIdx.x;
  if (t >= total) return;
  int i    = (int)(t & 15);
  int lane = (int)((t >> 4) & 31);
  long rest = t >> 9;
  int kk32 = K >> 5;
  int kk = (int)(rest % kk32);
  int tm = (int)(rest / kk32);
  int l = lane & 15, h = lane >> 4;
  int k = kk * 32 + h * 8 + ((i < 8) ? i : (8 + i));
  int m = tm * 16 + l;
  Ap[t] = (_Float16)A[(size_t)m * K + k];
}

__global__ void pack_w_kernel(const float* __restrict__ W,
                              _Float16* __restrict__ Wp,
                              int K, int N, long total) {
  long t = (long)blockIdx.x * blockDim.x + threadIdx.x;
  if (t >= total) return;
  int i    = (int)(t & 15);
  int lane = (int)((t >> 4) & 31);
  long rest = t >> 9;
  int kk32 = K >> 5;
  int kk = (int)(rest % kk32);
  int tn = (int)(rest / kk32);
  int l = lane & 15, h = lane >> 4;
  int k = kk * 32 + h * 8 + ((i < 8) ? i : (8 + i));
  int n = tn * 16 + l;
  Wp[t] = (_Float16)W[(size_t)k * N + n];
}

// ---------------------------------------------------------------------------
// WMMA GEMM on packed operands: C[M,N] = A @ W + bias.
// One wave -> 16x64 output tile (4 accumulators). Inner k-step:
//   1x v16h A load + 4x v16h B loads + 4x v_wmma_f32_16x16x32_f16.
// act: 0 none, 1 GELU.  rowmask: optional per-row zeroing (leader pad_mask).
// ---------------------------------------------------------------------------
__global__ void gemm_kernel(const _Float16* __restrict__ Ap,
                            const _Float16* __restrict__ Wp,
                            const float* __restrict__ bias,
                            float* __restrict__ C,
                            int M, int K, int N, int ntn4,
                            int act, const int* __restrict__ rowmask) {
  int wid  = (int)((blockIdx.x * blockDim.x + threadIdx.x) >> 5);
  int lane = threadIdx.x & 31;
  int tm  = wid / ntn4;
  int tnb = (wid % ntn4) * 4;
  if (tm * 16 >= M) return;
  int kk32 = K >> 5;
  const _Float16* ab = Ap + ((size_t)tm * kk32) * 512 + (size_t)lane * 16;
  const _Float16* wb = Wp + ((size_t)tnb * kk32) * 512 + (size_t)lane * 16;
  size_t wstride = (size_t)kk32 * 512;

  v8f acc[4] = {};
  for (int kk = 0; kk < kk32; ++kk) {
    v16h af = *(const v16h*)(ab + (size_t)kk * 512);
#pragma unroll
    for (int j = 0; j < 4; ++j) {
      v16h bf = *(const v16h*)(wb + (size_t)j * wstride + (size_t)kk * 512);
      acc[j] = __builtin_amdgcn_wmma_f32_16x16x32_f16(false, af, false, bf,
                                                      (short)0, acc[j],
                                                      false, false);
    }
  }
  // D layout: lane<16 -> rows 0..7, lane>=16 -> rows 8..15
  int l = lane & 15, h = lane >> 4;
  int mbase = tm * 16 + h * 8;
#pragma unroll
  for (int j = 0; j < 4; ++j) {
    int n = (tnb + j) * 16 + l;
    float bv = bias ? bias[n] : 0.0f;
#pragma unroll
    for (int r = 0; r < 8; ++r) {
      int m = mbase + r;
      float v = acc[j][r] + bv;
      if (act == 1) v = gelu_f(v);
      if (rowmask && rowmask[m] == 0) v = 0.0f;
      C[(size_t)m * N + n] = v;
    }
  }
}

// ---------------------------------------------------------------------------
// In-place LayerNorm + GELU, one wave per row, N % 32 == 0
// ---------------------------------------------------------------------------
__global__ void ln_gelu_kernel(float* __restrict__ X,
                               const float* __restrict__ g,
                               const float* __restrict__ beta,
                               int M, int N) {
  int row  = blockIdx.x * (blockDim.x >> 5) + (threadIdx.x >> 5);
  int lane = threadIdx.x & 31;
  if (row >= M) return;
  float* x = X + (size_t)row * N;
  int per = N >> 5;
  float s = 0.0f;
  for (int i = 0; i < per; ++i) s += x[lane + i * 32];
  s = wave_sum(s);
  float mu = s / (float)N;
  float s2 = 0.0f;
  for (int i = 0; i < per; ++i) {
    float v = x[lane + i * 32] - mu;
    s2 += v * v;
  }
  s2 = wave_sum(s2);
  float inv = rsqrtf(s2 / (float)N + 1e-5f);
  for (int i = 0; i < per; ++i) {
    int j = lane + i * 32;
    float v = (x[j] - mu) * inv * g[j] + beta[j];
    x[j] = gelu_f(v);
  }
}

// history.mean(axis=2): [B,A,8,32] -> [B,A,32]
__global__ void hist_mean_kernel(const float* __restrict__ h,
                                 float* __restrict__ out) {
  int t = blockIdx.x * blockDim.x + threadIdx.x;
  if (t >= M8 * 32) return;
  int c = t & 31, token = t >> 5;
  float s = 0.0f;
#pragma unroll
  for (int w = 0; w < HW; ++w) s += h[((size_t)token * HW + w) * 32 + c];
  out[t] = s * (1.0f / HW);
}

// kNN over positions: per (b,i) thread, top-16 nearest + softmax(-dist)
__global__ void knn_kernel(const float* __restrict__ pos,
                           int* __restrict__ nbidx,
                           float* __restrict__ memb) {
  int t = blockIdx.x * blockDim.x + threadIdx.x;
  if (t >= M8) return;
  int b = t / AA, i = t % AA;
  const float* pb = pos + (size_t)b * AA * 2;
  float px = pb[i * 2], py = pb[i * 2 + 1];
  float bd[KN]; int bi[KN];
#pragma unroll
  for (int k = 0; k < KN; ++k) { bd[k] = 1e30f; bi[k] = 0; }
  for (int j = 0; j < AA; ++j) {
    if (j == i) continue;
    float dx = px - pb[j * 2], dy = py - pb[j * 2 + 1];
    float d = sqrtf(fmaxf(dx * dx + dy * dy, 1e-12f));
    if (d < bd[KN - 1]) {
      int p = KN - 1;
      while (p > 0 && bd[p - 1] > d) { bd[p] = bd[p - 1]; bi[p] = bi[p - 1]; --p; }
      bd[p] = d; bi[p] = j;
    }
  }
  float sum = 0.0f; float e[KN];
#pragma unroll
  for (int k = 0; k < KN; ++k) { e[k] = expf(bd[0] - bd[k]); sum += e[k]; }
  float invs = 1.0f / sum;
#pragma unroll
  for (int k = 0; k < KN; ++k) {
    nbidx[(size_t)t * KN + k] = bi[k];
    memb[(size_t)t * KN + k]  = e[k] * invs;
  }
}

// Fused per-edge trust MLP: sigmoid(w2 . GELU(LN(oeW[i] + tW[nb] + b1)) + b2)
__global__ void trust_fuse_kernel(const float* __restrict__ oeW,
                                  const float* __restrict__ tW,
                                  const int* __restrict__ nb,
                                  const float* __restrict__ b1,
                                  const float* __restrict__ g,
                                  const float* __restrict__ beta,
                                  const float* __restrict__ w2,
                                  const float* __restrict__ b2,
                                  float* __restrict__ trust) {
  int e = blockIdx.x * (blockDim.x >> 5) + (threadIdx.x >> 5);
  if (e >= M8 * KN) return;
  int lane  = threadIdx.x & 31;
  int token = e / KN;
  int b     = token / AA;
  int j     = nb[e];
  const float* xr = oeW + (size_t)token * 128;
  const float* tr = tW + ((size_t)(b * AA + j)) * 128;
  float v[4]; float s = 0.0f;
#pragma unroll
  for (int i = 0; i < 4; ++i) {
    int c = lane + i * 32;
    v[i] = xr[c] + tr[c] + b1[c];
    s += v[i];
  }
  s = wave_sum(s);
  float mu = s * (1.0f / 128.0f);
  float s2 = 0.0f;
#pragma unroll
  for (int i = 0; i < 4; ++i) { float d = v[i] - mu; s2 += d * d; }
  s2 = wave_sum(s2);
  float inv = rsqrtf(s2 * (1.0f / 128.0f) + 1e-5f);
  float acc = 0.0f;
#pragma unroll
  for (int i = 0; i < 4; ++i) {
    int c = lane + i * 32;
    float u = (v[i] - mu) * inv * g[c] + beta[c];
    acc += gelu_f(u) * w2[c];
  }
  acc = wave_sum(acc);
  if (lane == 0) trust[e] = sigmoid_f(acc + b2[0]);
}

// edge weights: ew = memb*trust / (sum + eps)
__global__ void ew_kernel(const float* __restrict__ memb,
                          const float* __restrict__ trust,
                          float* __restrict__ ew) {
  int t = blockIdx.x * blockDim.x + threadIdx.x;
  if (t >= M8) return;
  float e[KN]; float s = 0.0f;
#pragma unroll
  for (int k = 0; k < KN; ++k) {
    e[k] = memb[(size_t)t * KN + k] * trust[(size_t)t * KN + k];
    s += e[k];
  }
  float inv = 1.0f / (s + EPSW);
#pragma unroll
  for (int k = 0; k < KN; ++k) ew[(size_t)t * KN + k] = e[k] * inv;
}

// small-N head: y[m, 0..N) = x[m,:] . W + b (optional sigmoid), wave per row
template <int NOUT>
__global__ void head_kernel(const float* __restrict__ X,
                            const float* __restrict__ W,
                            const float* __restrict__ bias,
                            float* __restrict__ Y,
                            int M, int Kd, int act) {
  int row  = blockIdx.x * (blockDim.x >> 5) + (threadIdx.x >> 5);
  int lane = threadIdx.x & 31;
  if (row >= M) return;
  const float* x = X + (size_t)row * Kd;
  float acc[NOUT];
#pragma unroll
  for (int n = 0; n < NOUT; ++n) acc[n] = 0.0f;
  for (int k = lane; k < Kd; k += 32) {
    float xv = x[k];
    const float* wr = W + (size_t)k * NOUT;
#pragma unroll
    for (int n = 0; n < NOUT; ++n) acc[n] += xv * wr[n];
  }
#pragma unroll
  for (int n = 0; n < NOUT; ++n) acc[n] = wave_sum(acc[n]);
  if (lane == 0) {
#pragma unroll
    for (int n = 0; n < NOUT; ++n) {
      float v = acc[n] + bias[n];
      if (act == 2) v = sigmoid_f(v);
      Y[(size_t)row * NOUT + n] = v;
    }
  }
}

// leader election per batch: local max over kNN -> mask, LDS scan -> slots
__global__ void leader_elect_kernel(const float* __restrict__ pot,
                                    const int* __restrict__ nbidx,
                                    int* __restrict__ lmask,
                                    int* __restrict__ a2s,
                                    int* __restrict__ lidx,
                                    int* __restrict__ padm) {
  int b = blockIdx.x;
  int i = threadIdx.x;           // AA threads
  __shared__ int s[AA];
  __shared__ int total;
  float p = pot[b * AA + i];
  int m = 1;
  for (int k = 0; k < KN; ++k) {
    int j = nbidx[((size_t)(b * AA + i)) * KN + k];
    if (p < pot[b * AA + j]) m = 0;
  }
  lmask[b * AA + i] = m;
  s[i] = m;
  __syncthreads();
  for (int off = 1; off < AA; off <<= 1) {
    int v = s[i];
    int add = (i >= off) ? s[i - off] : 0;
    __syncthreads();
    s[i] = v + add;
    __syncthreads();
  }
  int incl = s[i];
  if (i == AA - 1) total = incl;
  lidx[b * AA + i] = -1;
  __syncthreads();
  if (m) lidx[b * AA + (incl - 1)] = i;
  a2s[b * AA + i] = m ? (incl - 1) : -1;
  __syncthreads();
  padm[b * AA + i] = (i < total) ? 1 : 0;
}

// gather leader features / positions into slot order (masked)
__global__ void gather_leader_kernel(const float* __restrict__ feat,
                                     const float* __restrict__ pos,
                                     const int* __restrict__ lidx,
                                     const int* __restrict__ padm,
                                     float* __restrict__ leadf,
                                     float* __restrict__ leadp) {
  int t = blockIdx.x * blockDim.x + threadIdx.x;
  if (t >= M8 * 128) return;
  int f = t & 127, token = t >> 7, b = token >> 8;
  int li = lidx[token];
  int pm = padm[token];
  int sli = li < 0 ? 0 : li;
  int src = b * AA + sli;
  leadf[t] = pm ? feat[(size_t)src * 128 + f] : 0.0f;
  if (f < 2)
    leadp[(size_t)token * 2 + f] = pm ? pos[(size_t)src * 2 + f] : 0.0f;
}

// leader kNN (LK=8) with validity mask, softmax(-dist) over valid leaders
__global__ void leader_knn_kernel(const float* __restrict__ leadp,
                                  const int* __restrict__ padm,
                                  int* __restrict__ lnb,
                                  float* __restrict__ lw) {
  int t = blockIdx.x * blockDim.x + threadIdx.x;
  if (t >= M8) return;
  int b = t / AA, slot = t % AA;
  float bd[LK]; int bi[LK];
#pragma unroll
  for (int k = 0; k < LK; ++k) { bd[k] = 1e30f; bi[k] = 0; }
  if (padm[t]) {
    float px = leadp[(size_t)t * 2], py = leadp[(size_t)t * 2 + 1];
    for (int j = 0; j < AA; ++j) {
      if (j == slot) continue;
      if (!padm[b * AA + j]) continue;
      float dx = px - leadp[((size_t)(b * AA + j)) * 2];
      float dy = py - leadp[((size_t)(b * AA + j)) * 2 + 1];
      float d = sqrtf(fmaxf(dx * dx + dy * dy, 1e-12f));
      if (d < bd[LK - 1]) {
        int p = LK - 1;
        while (p > 0 && bd[p - 1] > d) { bd[p] = bd[p - 1]; bi[p] = bi[p - 1]; --p; }
        bd[p] = d; bi[p] = j;
      }
    }
  }
  float m0 = (bd[0] < 1e29f) ? bd[0] : 0.0f;
  float e[LK]; float sum = 0.0f;
#pragma unroll
  for (int k = 0; k < LK; ++k) {
    e[k] = (bd[k] < 1e29f) ? expf(m0 - bd[k]) : 0.0f;
    sum += e[k];
  }
  float inv = 1.0f / (sum + EPSW);
#pragma unroll
  for (int k = 0; k < LK; ++k) {
    lw[(size_t)t * LK + k]  = e[k] * inv;
    lnb[(size_t)t * LK + k] = (bd[k] < 1e29f) ? bi[k] : 0;
  }
}

// msg[b,i,f] = sum_k ew[b,i,k] * m[b, nb[b,i,k], f]   (linearity-hoisted)
__global__ void agg_intra_kernel(const float* __restrict__ m,
                                 const float* __restrict__ ew,
                                 const int* __restrict__ nb,
                                 float* __restrict__ out) {
  int t = blockIdx.x * blockDim.x + threadIdx.x;
  if (t >= M8 * 128) return;
  int f = t & 127, token = t >> 7, b = token >> 8;
  float acc = 0.0f;
#pragma unroll
  for (int k = 0; k < KN; ++k) {
    int j = nb[(size_t)token * KN + k];
    acc += ew[(size_t)token * KN + k] * m[(((size_t)(b * AA + j)) << 7) + f];
  }
  out[t] = acc;
}

__global__ void agg_inter_kernel(const float* __restrict__ m,
                                 const float* __restrict__ lw,
                                 const int* __restrict__ lnb,
                                 float* __restrict__ out) {
  int t = blockIdx.x * blockDim.x + threadIdx.x;
  if (t >= M8 * 128) return;
  int f = t & 127, token = t >> 7, b = token >> 8;
  float acc = 0.0f;
#pragma unroll
  for (int k = 0; k < LK; ++k) {
    int s2 = lnb[(size_t)token * LK + k];
    acc += lw[(size_t)token * LK + k] * m[(((size_t)(b * AA + s2)) << 7) + f];
  }
  out[t] = acc;
}

// broadcast leader latents back to agents via agent->slot map
__global__ void broadcast_kernel(const float* __restrict__ hl,
                                 const float* __restrict__ ew,
                                 const int* __restrict__ nb,
                                 const int* __restrict__ a2s,
                                 float* __restrict__ out) {
  int t = blockIdx.x * blockDim.x + threadIdx.x;
  if (t >= M8 * 128) return;
  int f = t & 127, token = t >> 7, b = token >> 8;
  float acc = 0.0f;
#pragma unroll
  for (int k = 0; k < KN; ++k) {
    int j = nb[(size_t)token * KN + k];
    int slot = a2s[b * AA + j];
    if (slot >= 0)
      acc += ew[(size_t)token * KN + k] * hl[(((size_t)(b * AA + slot)) << 7) + f];
  }
  out[t] = acc;
}

__global__ void concat2_kernel(const float* __restrict__ a, int na,
                               const float* __restrict__ b2, int nb2,
                               float* __restrict__ out) {
  int w = na + nb2;
  int t = blockIdx.x * blockDim.x + threadIdx.x;
  if (t >= M8 * w) return;
  int c = t % w, row = t / w;
  out[t] = (c < na) ? a[(size_t)row * na + c] : b2[(size_t)row * nb2 + (c - na)];
}

__global__ void concat3_kernel(const float* __restrict__ a, int na,
                               const float* __restrict__ b2, int nb2,
                               const float* __restrict__ c3, int nc,
                               float* __restrict__ out) {
  int w = na + nb2 + nc;
  int t = blockIdx.x * blockDim.x + threadIdx.x;
  if (t >= M8 * w) return;
  int c = t % w, row = t / w;
  float v;
  if (c < na) v = a[(size_t)row * na + c];
  else if (c < na + nb2) v = b2[(size_t)row * nb2 + (c - na)];
  else v = c3[(size_t)row * nc + (c - na - nb2)];
  out[t] = v;
}

__global__ void copy_f_kernel(const float* __restrict__ s,
                              float* __restrict__ d, long n) {
  long t = (long)blockIdx.x * blockDim.x + threadIdx.x;
  if (t < n) d[t] = s[t];
}

// ---------------------------------------------------------------------------
// host orchestration
// ---------------------------------------------------------------------------
extern "C" void kernel_launch(void* const* d_in, const int* in_sizes, int n_in,
                              void* d_out, int out_size, void* d_ws, size_t ws_size,
                              hipStream_t stream) {
  (void)in_sizes; (void)n_in; (void)out_size; (void)ws_size;

  auto F = [&](int i) -> const float* { return (const float*)d_in[i]; };
  const float* obs  = F(0);   // [B,A,64]
  const float* pos  = F(1);   // [B,A,2]
  const float* hist = F(2);   // [B,A,8,32]

  // workspace bump allocator (256B aligned)
  char* base = (char*)d_ws;
  size_t cur = 0;
  auto bump = [&](size_t bytes) -> void* {
    void* p = (void*)(base + cur);
    cur += (bytes + 255) & ~(size_t)255;
    return p;
  };
  auto af = [&](size_t n) -> float*    { return (float*)bump(n * 4); };
  auto ai = [&](size_t n) -> int*      { return (int*)bump(n * 4); };
  auto ah = [&](size_t n) -> _Float16* { return (_Float16*)bump(n * 2); };

  // f32 buffers
  float* a32    = af((size_t)M8 * 32);
  float* b128h  = af((size_t)M8 * 128);   // reused: beh hidden, then pot hidden
  float* beh    = af((size_t)M8 * 64);
  float* t256a  = af((size_t)M8 * 256);
  float* t256b  = af((size_t)M8 * 256);
  float* t320   = af((size_t)M8 * 320);
  float* oe     = af((size_t)M8 * 128);
  float* oeW    = af((size_t)M8 * 128);
  float* tW     = af((size_t)M8 * 128);
  float* catob  = af((size_t)M8 * 192);
  float* feat   = af((size_t)M8 * 128);
  float* pot    = af((size_t)M8);
  float* memb   = af((size_t)M8 * KN);
  float* trustb = af((size_t)M8 * KN);
  float* ew     = af((size_t)M8 * KN);
  float* leadf  = af((size_t)M8 * 128);
  float* leadp  = af((size_t)M8 * 2);
  float* lw     = af((size_t)M8 * LK);
  float* hbuf   = af((size_t)M8 * 128);
  float* mtmp   = af((size_t)M8 * 128);
  float* magg   = af((size_t)M8 * 128);
  float* hl     = af((size_t)M8 * 128);
  float* bcast  = af((size_t)M8 * 128);
  // int buffers
  int* nbidx = ai((size_t)M8 * KN);
  int* lmask = ai((size_t)M8);
  int* a2s   = ai((size_t)M8);
  int* lidx  = ai((size_t)M8);
  int* padm  = ai((size_t)M8);
  int* lnb   = ai((size_t)M8 * LK);
  // f16 packed-activation staging (max K = 320)
  _Float16* X16 = ah((size_t)M8 * 320);

  // pack one weight matrix into WMMA fragment order, up front
  auto cw = [&](int pidx, int K, int N) -> const _Float16* {
    size_t n = (size_t)K * N;
    _Float16* p = ah(n);
    pack_w_kernel<<<(int)((n + 255) / 256), 256, 0, stream>>>(F(pidx), p, K, N, (long)n);
    return p;
  };

  // weights -> packed f16 (param flattening: dict insertion; per layer W,b[,g,beta])
  const _Float16* w_obs0  = cw(3, 64, 256);
  const _Float16* w_obs1  = cw(7, 256, 256);
  const _Float16* w_obs2  = cw(11, 256, 128);
  const _Float16* w_beh0  = cw(13, 32, 128);
  const _Float16* w_beh1  = cw(17, 128, 64);
  // trust W0 [192,128]: pack top 128 rows (oe part) and bottom 64 rows (beh
  // part) as two independent packed matrices (linearity split).
  const _Float16* w_tr_t;
  const _Float16* w_tr_b;
  {
    size_t nt = (size_t)128 * 128;
    _Float16* p = ah(nt);
    pack_w_kernel<<<(int)((nt + 255) / 256), 256, 0, stream>>>(F(19), p, 128, 128, (long)nt);
    // NOTE: top block rows 0..127 of W0 are contiguous [128,128] row-major
    w_tr_t = p;
    size_t nb2 = (size_t)64 * 128;
    _Float16* q = ah(nb2);
    pack_w_kernel<<<(int)((nb2 + 255) / 256), 256, 0, stream>>>(
        F(19) + (size_t)128 * 128, q, 64, 128, (long)nb2);
    w_tr_b = q;
  }
  const _Float16* w_feat0 = cw(25, 192, 256);
  const _Float16* w_feat1 = cw(29, 256, 128);
  const _Float16* w_lead0 = cw(31, 128, 128);
  const _Float16* w_msga  = cw(37, 128, 128);
  const _Float16* w_upda  = cw(39, 256, 128);
  const _Float16* w_msgl  = cw(41, 128, 128);
  const _Float16* w_updl  = cw(43, 256, 128);
  const _Float16* w_act0  = cw(45, 320, 256);
  const _Float16* w_act1  = cw(49, 256, 256);
  const _Float16* w_cri0  = cw(55, 320, 256);
  const _Float16* w_cri1  = cw(59, 256, 256);

  auto gemm = [&](const float* A, int K, int N, const _Float16* W,
                  const float* bias, float* C, int act, const int* rowmask) {
    long an = (long)M8 * K;
    pack_a_kernel<<<(int)((an + 255) / 256), 256, 0, stream>>>(A, X16, K, an);
    int ntn4 = N / 64;                       // 4 column tiles per wave
    int waves = (M8 / 16) * ntn4;
    gemm_kernel<<<(waves + 7) / 8, 256, 0, stream>>>(X16, W, bias, C,
                                                     M8, K, N, ntn4, act, rowmask);
  };
  auto ln = [&](float* X, int pg, int pb, int N) {
    ln_gelu_kernel<<<M8 / 8, 256, 0, stream>>>(X, F(pg), F(pb), M8, N);
  };

  // 1) behavior encoder: mean(history) -> 32->128 (LN+GELU) -> 64
  hist_mean_kernel<<<(M8 * 32 + 255) / 256, 256, 0, stream>>>(hist, a32);
  gemm(a32, 32, 128, w_beh0, F(14), b128h, 0, nullptr);
  ln(b128h, 15, 16, 128);
  gemm(b128h, 128, 64, w_beh1, F(18), beh, 0, nullptr);

  // 2) obs encoder: 64->256->256->128
  gemm(obs, 64, 256, w_obs0, F(4), t256a, 0, nullptr);
  ln(t256a, 5, 6, 256);
  gemm(t256a, 256, 256, w_obs1, F(8), t256b, 0, nullptr);
  ln(t256b, 9, 10, 256);
  gemm(t256b, 256, 128, w_obs2, F(12), oe, 0, nullptr);

  // 3) kNN + membership softmax
  knn_kernel<<<M8 / 256, 256, 0, stream>>>(pos, nbidx, memb);

  // 4) trust: split first linear by linearity (avoids 100 MB gather tensor)
  gemm(oe, 128, 128, w_tr_t, nullptr, oeW, 0, nullptr);
  gemm(beh, 64, 128, w_tr_b, nullptr, tW, 0, nullptr);
  trust_fuse_kernel<<<(M8 * KN) / 8, 256, 0, stream>>>(
      oeW, tW, nbidx, F(20), F(21), F(22), F(23), F(24), trustb);
  ew_kernel<<<M8 / 256, 256, 0, stream>>>(memb, trustb, ew);

  // 5) feat_proj: concat(oe,beh) -> 192->256 (LN+GELU) ->128
  concat2_kernel<<<(M8 * 192 + 255) / 256, 256, 0, stream>>>(oe, 128, beh, 64, catob);
  gemm(catob, 192, 256, w_feat0, F(26), t256a, 0, nullptr);
  ln(t256a, 27, 28, 256);
  gemm(t256a, 256, 128, w_feat1, F(30), feat, 0, nullptr);

  // 6) leader potential: 128->128 (LN+GELU) -> 1, sigmoid
  gemm(feat, 128, 128, w_lead0, F(32), b128h, 0, nullptr);
  ln(b128h, 33, 34, 128);
  head_kernel<1><<<M8 / 8, 256, 0, stream>>>(b128h, F(35), F(36), pot, M8, 128, 2);

  // 7) leader election + slot gathers + leader kNN
  leader_elect_kernel<<<BB, AA, 0, stream>>>(pot, nbidx, lmask, a2s, lidx, padm);
  gather_leader_kernel<<<(M8 * 128 + 255) / 256, 256, 0, stream>>>(
      feat, pos, lidx, padm, leadf, leadp);
  leader_knn_kernel<<<M8 / 256, 256, 0, stream>>>(leadp, padm, lnb, lw);

  // 8) intra-group consensus x2 (msg linear hoisted before gather)
  copy_f_kernel<<<(M8 * 128 + 255) / 256, 256, 0, stream>>>(feat, hbuf, (long)M8 * 128);
  for (int it = 0; it < 2; ++it) {
    gemm(hbuf, 128, 128, w_msga, F(38), mtmp, 0, nullptr);
    agg_intra_kernel<<<(M8 * 128 + 255) / 256, 256, 0, stream>>>(mtmp, ew, nbidx, magg);
    concat2_kernel<<<(M8 * 256 + 255) / 256, 256, 0, stream>>>(hbuf, 128, magg, 128, t256a);
    gemm(t256a, 256, 128, w_upda, F(40), hbuf, 1, nullptr);
  }

  // 9) inter-leader consensus x2 (masked rows)
  copy_f_kernel<<<(M8 * 128 + 255) / 256, 256, 0, stream>>>(leadf, hl, (long)M8 * 128);
  for (int it = 0; it < 2; ++it) {
    gemm(hl, 128, 128, w_msgl, F(42), mtmp, 0, nullptr);
    agg_inter_kernel<<<(M8 * 128 + 255) / 256, 256, 0, stream>>>(mtmp, lw, lnb, magg);
    concat2_kernel<<<(M8 * 256 + 255) / 256, 256, 0, stream>>>(hl, 128, magg, 128, t256a);
    gemm(t256a, 256, 128, w_updl, F(44), hl, 1, padm);
  }

  // 10) broadcast leaders back, fuse, actor/critic heads
  broadcast_kernel<<<(M8 * 128 + 255) / 256, 256, 0, stream>>>(hl, ew, nbidx, a2s, bcast);
  concat3_kernel<<<(M8 * 320 + 255) / 256, 256, 0, stream>>>(
      hbuf, 128, bcast, 128, beh, 64, t320);

  float* out = (float*)d_out;
  float* out_logits = out;                       // [B,A,10]  = 81920
  float* out_values = out + 81920;               // [B,A]     = 8192
  float* out_ew     = out + 81920 + 8192;        // [B,A,16]  = 131072
  float* out_pot    = out + 81920 + 8192 + 131072;

  gemm(t320, 320, 256, w_act0, F(46), t256a, 0, nullptr);
  ln(t256a, 47, 48, 256);
  gemm(t256a, 256, 256, w_act1, F(50), t256b, 0, nullptr);
  ln(t256b, 51, 52, 256);
  head_kernel<10><<<M8 / 8, 256, 0, stream>>>(t256b, F(53), F(54), out_logits, M8, 256, 0);

  gemm(t320, 320, 256, w_cri0, F(56), t256a, 0, nullptr);
  ln(t256a, 57, 58, 256);
  gemm(t256a, 256, 256, w_cri1, F(60), t256b, 0, nullptr);
  ln(t256b, 61, 62, 256);
  head_kernel<1><<<M8 / 8, 256, 0, stream>>>(t256b, F(63), F(64), out_values, M8, 256, 0);

  copy_f_kernel<<<(M8 * KN + 255) / 256, 256, 0, stream>>>(ew, out_ew, (long)M8 * KN);
  copy_f_kernel<<<(M8 + 255) / 256, 256, 0, stream>>>(pot, out_pot, (long)M8);
}
